// RingDilatedAttentionHilbertOptimizedFixedV2_64690797412680
// MI455X (gfx1250) — matrix operand order repaired and checked by reading, not compiled, not run on hardware
//
#include <hip/hip_runtime.h>

// ---------------------------------------------------------------------------
// Multi-head attention (B=1, S=4096, E=768, H=12, D=64) for MI455X (gfx1250).
// All three matmul stages run on v_wmma_f32_16x16x32_bf16 (wave32 WMMA).
// Flash attention: 8 waves/block share one head; K and V^T chunks are staged
// in double-buffered LDS (one b128 per thread per buffer) so K/V L2 traffic
// is amortized 8x; online softmax is per-lane scalar via transposed scores.
// ---------------------------------------------------------------------------

#define S_LEN 4096
#define EMB   768
#define NH    12
#define HD    64
#define N3    2304   // 3*EMB

typedef __attribute__((ext_vector_type(16))) __bf16       bf16x16;
typedef __attribute__((ext_vector_type(8)))  float        floatx8;
typedef __attribute__((ext_vector_type(4)))  unsigned int uint4v;

union AFrag {            // 32 bytes: one WMMA A or B operand (16 bf16)
  uint4v  chunk[2];
  bf16x16 v;
};
union PFrag {            // built element-wise from shuffled probabilities
  unsigned short s[16];
  bf16x16        v;
};

#define F8_ZERO ((floatx8){0.f,0.f,0.f,0.f,0.f,0.f,0.f,0.f})

__device__ __forceinline__ unsigned short f2bf(float f) {
  unsigned int u = __builtin_bit_cast(unsigned int, f);
  u += 0x7fffu + ((u >> 16) & 1u);           // round-to-nearest-even
  return (unsigned short)(u >> 16);
}

// ---------------------------------------------------------------- converts
__global__ void cvt_f32_bf16(const float* __restrict__ in,
                             unsigned short* __restrict__ out, int n) {
  int i = blockIdx.x * blockDim.x + threadIdx.x;
  if (i < n) out[i] = f2bf(in[i]);
}

// in: [rows][cols] f32   ->   out: [cols][rows] bf16  (weight pre-transpose)
__global__ void transpose_bf16(const float* __restrict__ in,
                               unsigned short* __restrict__ out,
                               int rows, int cols) {
  int i = blockIdx.x * blockDim.x + threadIdx.x;
  if (i < rows * cols) {
    int c = i / rows;
    int r = i - c * rows;
    out[(size_t)c * rows + r] = f2bf(in[(size_t)r * cols + c]);
  }
}

// ------------------------------------------------------------- QKV GEMM
// C[S][N3] = xbf[S][EMB] @ wqkvT[N3][EMB]^T + bias, scattered into
// Q[h][s][d], K[h][s][d], V^T[h][d][s] (bf16). 64x64 tile per wave.
__global__ void __launch_bounds__(256)
qkv_gemm(const unsigned short* __restrict__ A,
         const unsigned short* __restrict__ BT,
         const float* __restrict__ bias,
         unsigned short* __restrict__ Qb,
         unsigned short* __restrict__ Kb,
         unsigned short* __restrict__ VTb) {
  const int lane = threadIdx.x & 31;
  const int wave = blockIdx.x * (blockDim.x >> 5) + (threadIdx.x >> 5);
  const int mWaves = S_LEN / 64;
  const int m0 = (wave % mWaves) * 64;
  const int n0 = (wave / mWaves) * 64;
  const int hl  = lane >> 4;
  const int l16 = lane & 15;

  floatx8 acc[4][4];
#pragma unroll
  for (int i = 0; i < 4; ++i)
#pragma unroll
    for (int j = 0; j < 4; ++j) acc[i][j] = F8_ZERO;

  for (int kk = 0; kk < EMB; kk += 32) {
    AFrag a[4];
#pragma unroll
    for (int i = 0; i < 4; ++i) {
      const unsigned short* ap =
          A + (size_t)(m0 + i * 16 + l16) * EMB + kk + 8 * hl;
      a[i].chunk[0] = *(const uint4v*)ap;
      a[i].chunk[1] = *(const uint4v*)(ap + 16);
    }
#pragma unroll
    for (int j = 0; j < 4; ++j) {
      AFrag b;
      const unsigned short* bp =
          BT + (size_t)(n0 + j * 16 + l16) * EMB + kk + 8 * hl;
      b.chunk[0] = *(const uint4v*)bp;
      b.chunk[1] = *(const uint4v*)(bp + 16);
#pragma unroll
      for (int i = 0; i < 4; ++i)
        acc[i][j] = __builtin_amdgcn_wmma_f32_16x16x32_bf16(
            false, a[i].v, false, b.v, (short)0, acc[i][j], false, false);
    }
  }

#pragma unroll
  for (int j = 0; j < 4; ++j) {
    int n = n0 + j * 16 + l16;
    float bv = bias[n];
    int which = n / EMB;               // 0=Q 1=K 2=V (uniform per 64-tile)
    int rem = n - which * EMB;
    int h = rem >> 6;
    int d = rem & 63;
#pragma unroll
    for (int i = 0; i < 4; ++i) {
#pragma unroll
      for (int r = 0; r < 8; ++r) {
        int s = m0 + i * 16 + r + 8 * hl;
        unsigned short bf = f2bf(acc[i][j][r] + bv);
        if (which == 0)      Qb[((size_t)h * S_LEN + s) * HD + d] = bf;
        else if (which == 1) Kb[((size_t)h * S_LEN + s) * HD + d] = bf;
        else                 VTb[((size_t)h * HD + d) * S_LEN + s] = bf;
      }
    }
  }
}

// --------------------------------------------------------- flash attention
// Block = 8 waves, one head, 8 consecutive 16-query tiles. K / V^T chunks
// (32 keys) staged in double-buffered LDS, shared by all 8 waves.
__global__ void __launch_bounds__(256)
flash_attn(const unsigned short* __restrict__ Qb,
           const unsigned short* __restrict__ Kb,
           const unsigned short* __restrict__ VTb,
           unsigned short* __restrict__ Ob) {
  __shared__ __attribute__((aligned(16))) unsigned short Ks[2][32 * HD];
  __shared__ __attribute__((aligned(16))) unsigned short Vs[2][HD * 32];

  const int tid   = threadIdx.x;
  const int lane  = tid & 31;
  const int h     = blockIdx.y;
  const int qtile = blockIdx.x * (blockDim.x >> 5) + (tid >> 5);
  const int qbase = qtile * 16;
  const int hl  = lane >> 4;
  const int l16 = lane & 15;

  const unsigned short* Qh = Qb + (size_t)h * S_LEN * HD;
  const unsigned short* Kh = Kb + (size_t)h * S_LEN * HD;
  const unsigned short* Vh = VTb + (size_t)h * HD * S_LEN;

  // Q as B-fragments (lane n = q, table k order along d), reused for all keys
  AFrag qf[2];
#pragma unroll
  for (int ds = 0; ds < 2; ++ds) {
    const unsigned short* qp =
        Qh + (size_t)(qbase + l16) * HD + ds * 32 + 8 * hl;
    qf[ds].chunk[0] = *(const uint4v*)qp;
    qf[ds].chunk[1] = *(const uint4v*)(qp + 16);
  }

  const float sc = 0.125f * 1.44269504088896f;  // 1/sqrt(64) * log2(e)

  floatx8 od[4];
#pragma unroll
  for (int t = 0; t < 4; ++t) od[t] = F8_ZERO;
  float mrow = -1e30f, lrow = 0.f;

  // cooperative staging: one b128 per thread per buffer
  const int vrow = tid >> 2;           // d row 0..63
  const int vcol = (tid & 3) * 8;      // 8-short slice within 32 keys
  uint4v kst = *(const uint4v*)(Kh + tid * 8);                 // chunk kb=0
  uint4v vst = *(const uint4v*)(Vh + (size_t)vrow * S_LEN + vcol);
  *(uint4v*)&Ks[0][tid * 8] = kst;
  *(uint4v*)&Vs[0][vrow * 32 + vcol] = vst;

  int buf = 0;
  for (int kb = 0; kb < S_LEN; kb += 32) {
    __syncthreads();                   // staged chunk visible to all waves
    const bool more = (kb + 32) < S_LEN;
    if (more) {                        // prefetch next chunk into registers
      kst = *(const uint4v*)(Kh + (size_t)(kb + 32) * HD + tid * 8);
      vst = *(const uint4v*)(Vh + (size_t)vrow * S_LEN + kb + 32 + vcol);
    }

    // S^T tiles (k_local rows, q cols) from LDS, scaled into log2 domain
    float st[2][8];
#pragma unroll
    for (int t = 0; t < 2; ++t) {
      floatx8 sacc = F8_ZERO;
#pragma unroll
      for (int ds = 0; ds < 2; ++ds) {
        AFrag kf;
        const unsigned short* kp =
            &Ks[buf][(t * 16 + l16) * HD + ds * 32 + 8 * hl];
        kf.chunk[0] = *(const uint4v*)kp;
        kf.chunk[1] = *(const uint4v*)(kp + 16);
        sacc = __builtin_amdgcn_wmma_f32_16x16x32_bf16(
            false, kf.v, false, qf[ds].v, (short)0, sacc, false, false);
      }
#pragma unroll
      for (int r = 0; r < 8; ++r) st[t][r] = sacc[r] * sc;
    }

    // online softmax (each lane = one query; pair lanes L,L+16 share q)
    float cmax = st[0][0];
#pragma unroll
    for (int t = 0; t < 2; ++t)
#pragma unroll
      for (int r = 0; r < 8; ++r) cmax = fmaxf(cmax, st[t][r]);
    cmax = fmaxf(cmax, __shfl_xor(cmax, 16, 32));
    float mnew  = fmaxf(mrow, cmax);
    float alpha = exp2f(mrow - mnew);
    lrow *= alpha;
#pragma unroll
    for (int t = 0; t < 4; ++t)
#pragma unroll
      for (int r = 0; r < 8; ++r) od[t][r] *= alpha;

    float p[2][8];
    float psum = 0.f;
#pragma unroll
    for (int t = 0; t < 2; ++t)
#pragma unroll
      for (int r = 0; r < 8; ++r) {
        p[t][r] = exp2f(st[t][r] - mnew);
        psum += p[t][r];
      }
    psum += __shfl_xor(psum, 16, 32);
    lrow += psum;
    mrow = mnew;

    // C-layout -> B-fragment layout for P^T via lane shuffles
    PFrag pb;
#pragma unroll
    for (int e = 0; e < 16; ++e) {
      int r = e & 7;
      int srcLane = l16 + ((e >> 3) << 4);
      float v0 = __shfl(p[0][r], srcLane, 32);
      float v1 = __shfl(p[1][r], srcLane, 32);
      pb.s[e] = f2bf(hl ? v1 : v0);
    }

    // O^T += V^T (A, from LDS) x P^T (B)
#pragma unroll
    for (int dt = 0; dt < 4; ++dt) {
      AFrag vf;
      const unsigned short* vp = &Vs[buf][(dt * 16 + l16) * 32 + 16 * hl];
      vf.chunk[0] = *(const uint4v*)vp;
      vf.chunk[1] = *(const uint4v*)(vp + 8);
      od[dt] = __builtin_amdgcn_wmma_f32_16x16x32_bf16(
          false, vf.v, false, pb.v, (short)0, od[dt], false, false);
    }

    if (more) {                        // store prefetch into the other buffer
      *(uint4v*)&Ks[buf ^ 1][tid * 8] = kst;
      *(uint4v*)&Vs[buf ^ 1][vrow * 32 + vcol] = vst;
    }
    buf ^= 1;
  }

  float rl = 1.0f / lrow;
  int q = qbase + l16;
#pragma unroll
  for (int dt = 0; dt < 4; ++dt)
#pragma unroll
    for (int r = 0; r < 8; ++r) {
      int d = dt * 16 + r + 8 * hl;
      Ob[(size_t)q * EMB + h * HD + d] = f2bf(od[dt][r] * rl);
    }
}

// ------------------------------------------------------------- out GEMM
// d_out[S][EMB] (f32) = attn[S][EMB] @ woutT[EMB][EMB]^T + b_out
__global__ void __launch_bounds__(256)
out_gemm(const unsigned short* __restrict__ A,
         const unsigned short* __restrict__ BT,
         const float* __restrict__ bias,
         float* __restrict__ C) {
  const int lane = threadIdx.x & 31;
  const int wave = blockIdx.x * (blockDim.x >> 5) + (threadIdx.x >> 5);
  const int mWaves = S_LEN / 64;
  const int m0 = (wave % mWaves) * 64;
  const int n0 = (wave / mWaves) * 64;
  const int hl  = lane >> 4;
  const int l16 = lane & 15;

  floatx8 acc[4][4];
#pragma unroll
  for (int i = 0; i < 4; ++i)
#pragma unroll
    for (int j = 0; j < 4; ++j) acc[i][j] = F8_ZERO;

  for (int kk = 0; kk < EMB; kk += 32) {
    AFrag a[4];
#pragma unroll
    for (int i = 0; i < 4; ++i) {
      const unsigned short* ap =
          A + (size_t)(m0 + i * 16 + l16) * EMB + kk + 8 * hl;
      a[i].chunk[0] = *(const uint4v*)ap;
      a[i].chunk[1] = *(const uint4v*)(ap + 16);
    }
#pragma unroll
    for (int j = 0; j < 4; ++j) {
      AFrag b;
      const unsigned short* bp =
          BT + (size_t)(n0 + j * 16 + l16) * EMB + kk + 8 * hl;
      b.chunk[0] = *(const uint4v*)bp;
      b.chunk[1] = *(const uint4v*)(bp + 16);
#pragma unroll
      for (int i = 0; i < 4; ++i)
        acc[i][j] = __builtin_amdgcn_wmma_f32_16x16x32_bf16(
            false, a[i].v, false, b.v, (short)0, acc[i][j], false, false);
    }
  }

#pragma unroll
  for (int j = 0; j < 4; ++j) {
    int n = n0 + j * 16 + l16;
    float bv = bias[n];
#pragma unroll
    for (int i = 0; i < 4; ++i)
#pragma unroll
      for (int r = 0; r < 8; ++r) {
        int s = m0 + i * 16 + r + 8 * hl;
        C[(size_t)s * EMB + n] = acc[i][j][r] + bv;
      }
  }
}

// ---------------------------------------------------------------- launcher
extern "C" void kernel_launch(void* const* d_in, const int* in_sizes, int n_in,
                              void* d_out, int out_size, void* d_ws,
                              size_t ws_size, hipStream_t stream) {
  const float* x     = (const float*)d_in[0];
  const float* w_qkv = (const float*)d_in[1];
  const float* b_qkv = (const float*)d_in[2];
  const float* w_out = (const float*)d_in[3];
  const float* b_out = (const float*)d_in[4];
  float* out = (float*)d_out;

  char* ws = (char*)d_ws;
  unsigned short* xbf   = (unsigned short*)(ws);              // 6291456 B
  unsigned short* wqkvT = (unsigned short*)(ws + 6291456);    // 3538944 B
  unsigned short* woutT = (unsigned short*)(ws + 9830400);    // 1179648 B
  unsigned short* Qb    = (unsigned short*)(ws + 11010048);   // 6291456 B
  unsigned short* Kb    = (unsigned short*)(ws + 17301504);   // 6291456 B
  unsigned short* VTb   = (unsigned short*)(ws + 23592960);   // 6291456 B
  unsigned short* attnb = (unsigned short*)(ws + 29884416);   // 6291456 B

  cvt_f32_bf16<<<(S_LEN * EMB + 255) / 256, 256, 0, stream>>>(x, xbf,
                                                              S_LEN * EMB);
  transpose_bf16<<<(EMB * N3 + 255) / 256, 256, 0, stream>>>(w_qkv, wqkvT,
                                                             EMB, N3);
  transpose_bf16<<<(EMB * EMB + 255) / 256, 256, 0, stream>>>(w_out, woutT,
                                                              EMB, EMB);

  qkv_gemm<<<(S_LEN / 64) * (N3 / 64) / 8, 256, 0, stream>>>(
      xbf, wqkvT, b_qkv, Qb, Kb, VTb);

  flash_attn<<<dim3(S_LEN / 16 / 8, NH), 256, 0, stream>>>(Qb, Kb, VTb, attnb);

  out_gemm<<<(S_LEN / 64) * (EMB / 64) / 8, 256, 0, stream>>>(attnb, woutT,
                                                              b_out, out);
}
